// GeniePath_86569360818439
// MI455X (gfx1250) — matrix operand order repaired and checked by reading, not compiled, not run on hardware
//
#include <hip/hip_runtime.h>
#include <hip/hip_bf16.h>
#include <math.h>

#define NN   100000
#define EE   3200000
#define IND  256
#define HID  16
#define OUTD 64
#define NLAYERS 2

typedef __attribute__((ext_vector_type(16))) _Float16 v16h;
typedef __attribute__((ext_vector_type(8)))  float    v8f;

// ---- ISA 7.12.2 fragment index helpers (wave32) ----
// 16-bit A 16x32: lane m=lane&15; vgpr v=j>>1; k = (v>=4)*16 + (lane>=16)*8 + (v&3)*2 + (j&1)
__device__ __forceinline__ int a_kidx(int lane, int j) {
  int v = j >> 1;
  return ((v >> 2) << 4) + (((lane >> 4) & 1) << 3) + ((v & 3) << 1) + (j & 1);
}
// 16-bit B 32x16: lane n=lane&15; k = (j>=8)*16 + (lane>=16)*8 + (j&7) (same as A)
__device__ __forceinline__ int b_kidx(int lane, int j) {
  return ((j >> 3) << 4) + (((lane >> 4) & 1) << 3) + (j & 7);
}
// f32 C/D 16x16: d[r] -> M = r + 8*(lane>=16), N = lane&15

// ---- order-preserving float<->uint map for atomic max on f32 ----
__device__ __forceinline__ unsigned fmap(float f) {
  unsigned u = __float_as_uint(f);
  return (u & 0x80000000u) ? ~u : (u | 0x80000000u);
}
__device__ __forceinline__ float funmap(unsigned m) {
  unsigned u = (m & 0x80000000u) ? (m ^ 0x80000000u) : ~m;
  return __uint_as_float(u);
}
#define NEG_INF_MAPPED 0x007FFFFFu   // fmap(-inf)

__device__ __forceinline__ float sigm(float x) { return 1.0f / (1.0f + expf(-x)); }

// Wave-local LDS transpose store of a 16x16 f32 WMMA D-tile into dst[N,16].
// In: acc per D layout. Out: 2 coalesced b128 stores per lane, one row guard.
// Per-wave LDS ops are in-order (DScnt), wave_barrier pins compiler ordering.
__device__ __forceinline__ void store_tile16(float* __restrict__ dstp, int base,
                                             const v8f& acc, float* __restrict__ sm) {
  int lane = threadIdx.x & 31;
  int n = lane & 15;
  int hi = (lane >> 4) & 1;
#pragma unroll
  for (int r = 0; r < 8; ++r) sm[(r + 8 * hi) * 16 + n] = acc[r];
  __builtin_amdgcn_wave_barrier();
  int row = lane >> 1;
  int off = (lane & 1) * 8;
  if (base + row < NN) {
    const float4* s4 = (const float4*)(sm + row * 16 + off);
    float4* g4 = (float4*)(dstp + (size_t)(base + row) * HID + off);
    g4[0] = s4[0];
    g4[1] = s4[1];
  }
}

// ============ lin1: xh[N,16] = x[N,256] @ w1^T + b1 (f16 WMMA, K-loop) ============
__global__ void k_lin1(const float* __restrict__ x, const float* __restrict__ w1,
                       const float* __restrict__ b1, float* __restrict__ xh) {
  __shared__ _Float16 w1s[HID * IND];     // 8 KB: whole weight, f16
  __shared__ float    stg[8 * 256];       // 8 KB: per-wave store staging
  for (int i = threadIdx.x; i < HID * IND; i += blockDim.x) w1s[i] = (_Float16)w1[i];
  __syncthreads();

  int lane = threadIdx.x & 31;
  int wave = threadIdx.x >> 5;
  int base = blockIdx.x * 128 + wave * 16;
  int n = lane & 15;
  int row = base + (lane & 15); if (row >= NN) row = NN - 1;  // EXEC all-1 for WMMA
  v8f acc = {};
  for (int k0 = 0; k0 < IND; k0 += 32) {
    v16h a, b;
#pragma unroll
    for (int j = 0; j < 16; ++j) {
      a[j] = (_Float16)x[(size_t)row * IND + k0 + a_kidx(lane, j)];
      b[j] = w1s[n * IND + k0 + b_kidx(lane, j)];
    }
    acc = __builtin_amdgcn_wmma_f32_16x16x32_f16(false, a, false, b, (short)0, acc, false, false);
  }
  float bias = b1[n];
#pragma unroll
  for (int r = 0; r < 8; ++r) acc[r] += bias;
  store_tile16(xh, base, acc, stg + wave * 256);
}

// ============ GAT feat: feat[N,16] = xin[N,16] @ fc^T (K padded 16->32) ============
__global__ void k_gat_feat(const float* __restrict__ xin, const float* __restrict__ fc,
                           float* __restrict__ feat) {
  __shared__ _Float16 fcs[HID * 32];      // [n][k] zero-padded to K=32
  __shared__ float    stg[8 * 256];
  for (int i = threadIdx.x; i < HID * 32; i += blockDim.x) {
    int o = i >> 5, k = i & 31;
    fcs[i] = (k < HID) ? (_Float16)fc[o * HID + k] : (_Float16)0.0f;
  }
  __syncthreads();

  int lane = threadIdx.x & 31;
  int wave = threadIdx.x >> 5;
  int base = blockIdx.x * 128 + wave * 16;
  int row = base + (lane & 15); if (row >= NN) row = NN - 1;
  int n = lane & 15;
  v16h a, b;
#pragma unroll
  for (int j = 0; j < 16; ++j) {
    int ka = a_kidx(lane, j);
    a[j] = (ka < HID) ? (_Float16)xin[(size_t)row * HID + ka] : (_Float16)0.0f;
    b[j] = fcs[n * 32 + b_kidx(lane, j)];
  }
  v8f z = {};
  v8f acc = __builtin_amdgcn_wmma_f32_16x16x32_f16(false, a, false, b, (short)0, z, false, false);
  store_tile16(feat, base, acc, stg + wave * 256);
}

// ============ el/er node dots ============
__global__ void k_el_er(const float* __restrict__ feat, const float* __restrict__ al,
                        const float* __restrict__ ar, float* __restrict__ el,
                        float* __restrict__ er) {
  int i = blockIdx.x * blockDim.x + threadIdx.x;
  if (i >= NN) return;
  float sl = 0.f, sr = 0.f;
#pragma unroll
  for (int j = 0; j < HID; ++j) {
    float f = feat[(size_t)i * HID + j];
    sl += f * al[j];
    sr += f * ar[j];
  }
  el[i] = sl;
  er[i] = sr;
}

// ============ per-layer accumulator init ============
__global__ void k_layer_init(unsigned* __restrict__ nmax, float* __restrict__ nsum,
                             float* __restrict__ rst) {
  int i = blockIdx.x * blockDim.x + threadIdx.x;
  if (i < NN * HID) rst[i] = 0.0f;
  if (i < NN) { nmax[i] = NEG_INF_MAPPED; nsum[i] = 0.0f; }
}

__global__ void k_zero2(float* __restrict__ a, float* __restrict__ b, int count) {
  int i = blockIdx.x * blockDim.x + threadIdx.x;
  if (i < count) { a[i] = 0.0f; b[i] = 0.0f; }
}

// ============ edge passes (gathers L2-resident; f32 atomics dominate) ============
__device__ __forceinline__ float edge_e(const float* el, const float* er, int s, int d) {
  float e = el[s] + er[d];
  return e > 0.0f ? e : 0.2f * e;   // leaky_relu(0.2)
}

__global__ void k_edge_max(const long long* __restrict__ src, const long long* __restrict__ dst,
                           const float* __restrict__ el, const float* __restrict__ er,
                           unsigned* __restrict__ nmax) {
  int i = blockIdx.x * blockDim.x + threadIdx.x;
  if (i >= EE) return;
  int s = (int)src[i], d = (int)dst[i];
  atomicMax(&nmax[d], fmap(edge_e(el, er, s, d)));
}

__global__ void k_edge_sum(const long long* __restrict__ src, const long long* __restrict__ dst,
                           const float* __restrict__ el, const float* __restrict__ er,
                           const unsigned* __restrict__ nmax, float* __restrict__ nsum) {
  int i = blockIdx.x * blockDim.x + threadIdx.x;
  if (i >= EE) return;
  int s = (int)src[i], d = (int)dst[i];
  float e = edge_e(el, er, s, d);
  atomicAdd(&nsum[d], expf(e - funmap(nmax[d])));
}

__global__ void k_edge_aggr(const long long* __restrict__ src, const long long* __restrict__ dst,
                            const float* __restrict__ el, const float* __restrict__ er,
                            const unsigned* __restrict__ nmax, const float* __restrict__ nsum,
                            const float* __restrict__ feat, float* __restrict__ rst) {
  int i = blockIdx.x * blockDim.x + threadIdx.x;
  if (i >= EE) return;
  int s = (int)src[i], d = (int)dst[i];
  float e = edge_e(el, er, s, d);
  float alpha = expf(e - funmap(nmax[d])) / nsum[d];
  const float4* fs = (const float4*)(feat + (size_t)s * HID);
#pragma unroll
  for (int q = 0; q < 4; ++q) {
    float4 v = fs[q];
    atomicAdd(&rst[d * HID + q * 4 + 0], alpha * v.x);
    atomicAdd(&rst[d * HID + q * 4 + 1], alpha * v.y);
    atomicAdd(&rst[d * HID + q * 4 + 2], alpha * v.z);
    atomicAdd(&rst[d * HID + q * 4 + 3], alpha * v.w);
  }
}

// ============ LSTM cell: A=[tanh(rst+bias) | h] 16x32, B=[wih|whh] per gate tile ============
__global__ void k_lstm(const float* __restrict__ rst, const float* __restrict__ gbias,
                       float* __restrict__ hbuf, float* __restrict__ cbuf,
                       const float* __restrict__ wih, const float* __restrict__ whh,
                       const float* __restrict__ bih, const float* __restrict__ bhh) {
  __shared__ _Float16 wcat[OUTD * 32];    // 4 KB: [gate_out][k], k<16->wih, k>=16->whh
  for (int i = threadIdx.x; i < OUTD * 32; i += blockDim.x) {
    int o = i >> 5, k = i & 31;
    float v = (k < HID) ? wih[o * HID + k] : whh[o * HID + (k - HID)];
    wcat[i] = (_Float16)v;
  }
  __syncthreads();

  int lane = threadIdx.x & 31;
  int wave = threadIdx.x >> 5;
  int base = blockIdx.x * 128 + wave * 16;
  int row = base + (lane & 15); if (row >= NN) row = NN - 1;
  int n = lane & 15;
  v16h a;
#pragma unroll
  for (int j = 0; j < 16; ++j) {
    int k = a_kidx(lane, j);
    float v = (k < HID) ? tanhf(rst[(size_t)row * HID + k] + gbias[k])  // g = tanh(rst+bias)
                        : hbuf[(size_t)row * HID + (k - HID)];
    a[j] = (_Float16)v;
  }
  v8f acc[4];
#pragma unroll
  for (int nt = 0; nt < 4; ++nt) {        // gate tiles i,f,g,o (4*HID == 64)
    int ng = nt * 16 + n;
    v16h b;
#pragma unroll
    for (int j = 0; j < 16; ++j) b[j] = wcat[ng * 32 + b_kidx(lane, j)];
    v8f z = {};
    acc[nt] = __builtin_amdgcn_wmma_f32_16x16x32_f16(false, a, false, b, (short)0, z, false, false);
  }
  float bi = bih[0 * 16 + n] + bhh[0 * 16 + n];
  float bf = bih[1 * 16 + n] + bhh[1 * 16 + n];
  float bg = bih[2 * 16 + n] + bhh[2 * 16 + n];
  float bo = bih[3 * 16 + n] + bhh[3 * 16 + n];
  int hi = (lane >> 4) & 1;
  if (base + 16 <= NN) {                  // uniform fast path: no per-element predication
#pragma unroll
    for (int r = 0; r < 8; ++r) {
      int rr = base + r + 8 * hi;
      float cv = cbuf[(size_t)rr * HID + n];
      float cn = sigm(acc[1][r] + bf) * cv + sigm(acc[0][r] + bi) * tanhf(acc[2][r] + bg);
      float hn = sigm(acc[3][r] + bo) * tanhf(cn);
      cbuf[(size_t)rr * HID + n] = cn;
      hbuf[(size_t)rr * HID + n] = hn;
    }
  } else {
#pragma unroll
    for (int r = 0; r < 8; ++r) {
      int rr = base + r + 8 * hi;
      if (rr < NN) {
        float cv = cbuf[(size_t)rr * HID + n];
        float cn = sigm(acc[1][r] + bf) * cv + sigm(acc[0][r] + bi) * tanhf(acc[2][r] + bg);
        float hn = sigm(acc[3][r] + bo) * tanhf(cn);
        cbuf[(size_t)rr * HID + n] = cn;
        hbuf[(size_t)rr * HID + n] = hn;
      }
    }
  }
}

// ============ lin2: out[N,64] = h[N,16] @ w2^T + b2 (K padded to 32) ============
__global__ void k_lin2(const float* __restrict__ hbuf, const float* __restrict__ w2,
                       const float* __restrict__ b2, float* __restrict__ out) {
  __shared__ _Float16 w2s[OUTD * 32];     // 4 KB, zero-padded K
  __shared__ float    stg[8 * 1024];      // 32 KB: per-wave 16x64 staging
  for (int i = threadIdx.x; i < OUTD * 32; i += blockDim.x) {
    int o = i >> 5, k = i & 31;
    w2s[i] = (k < HID) ? (_Float16)w2[o * HID + k] : (_Float16)0.0f;
  }
  __syncthreads();

  int lane = threadIdx.x & 31;
  int wave = threadIdx.x >> 5;
  int base = blockIdx.x * 128 + wave * 16;
  int row = base + (lane & 15); if (row >= NN) row = NN - 1;
  int n = lane & 15;
  v16h a;
#pragma unroll
  for (int j = 0; j < 16; ++j) {
    int k = a_kidx(lane, j);
    a[j] = (k < HID) ? (_Float16)hbuf[(size_t)row * HID + k] : (_Float16)0.0f;
  }
  v8f acc[4];
#pragma unroll
  for (int nt = 0; nt < 4; ++nt) {
    int ng = nt * 16 + n;
    v16h b;
#pragma unroll
    for (int j = 0; j < 16; ++j) b[j] = w2s[ng * 32 + b_kidx(lane, j)];
    v8f z = {};
    acc[nt] = __builtin_amdgcn_wmma_f32_16x16x32_f16(false, a, false, b, (short)0, z, false, false);
  }
  // LDS transpose -> coalesced b128 row stores (+bias)
  float* sm = stg + wave * 1024;
  int hi = (lane >> 4) & 1;
#pragma unroll
  for (int nt = 0; nt < 4; ++nt)
#pragma unroll
    for (int r = 0; r < 8; ++r)
      sm[(r + 8 * hi) * OUTD + nt * 16 + n] = acc[nt][r];
  __builtin_amdgcn_wave_barrier();
  int rw = lane >> 1;
  int off = (lane & 1) * 32;              // 32 floats per lane = 8x float4, coalesced
  if (base + rw < NN) {
    float* g = out + (size_t)(base + rw) * OUTD + off;
#pragma unroll
    for (int q = 0; q < 8; ++q) {
      float4 v = ((const float4*)(sm + rw * OUTD + off))[q];
      v.x += b2[off + q * 4 + 0];
      v.y += b2[off + q * 4 + 1];
      v.z += b2[off + q * 4 + 2];
      v.w += b2[off + q * 4 + 3];
      ((float4*)g)[q] = v;
    }
  }
}

extern "C" void kernel_launch(void* const* d_in, const int* in_sizes, int n_in,
                              void* d_out, int out_size, void* d_ws, size_t ws_size,
                              hipStream_t stream) {
  const float*     x        = (const float*)d_in[0];
  const long long* src      = (const long long*)d_in[1];
  const long long* dst      = (const long long*)d_in[2];
  const float*     w1       = (const float*)d_in[3];
  const float*     b1       = (const float*)d_in[4];
  const float*     gat_fc   = (const float*)d_in[5];   // [L,16,16]
  const float*     attn_l   = (const float*)d_in[6];   // [L,1,16]
  const float*     attn_r   = (const float*)d_in[7];
  const float*     gat_bias = (const float*)d_in[8];   // [L,16]
  const float*     wih      = (const float*)d_in[9];   // [L,64,16]
  const float*     whh      = (const float*)d_in[10];
  const float*     bih      = (const float*)d_in[11];  // [L,64]
  const float*     bhh      = (const float*)d_in[12];
  const float*     w2       = (const float*)d_in[13];  // [64,16]
  const float*     b2       = (const float*)d_in[14];
  float*           out      = (float*)d_out;

  float*    xh   = (float*)d_ws;          // [N,16]
  float*    hbuf = xh   + (size_t)NN * HID;
  float*    cbuf = hbuf + (size_t)NN * HID;
  float*    feat = cbuf + (size_t)NN * HID;
  float*    rst  = feat + (size_t)NN * HID;
  float*    el   = rst  + (size_t)NN * HID;
  float*    er   = el   + NN;
  float*    nsum = er   + NN;
  unsigned* nmax = (unsigned*)(nsum + NN);

  dim3 blk(256);
  int gridWave = (NN + 127) / 128;              // 8 waves/block, 16 rows/wave
  int gridNode = (NN + 255) / 256;
  int gridNH   = (NN * HID + 255) / 256;
  int gridEdge = (EE + 255) / 256;

  k_zero2<<<gridNH, blk, 0, stream>>>(hbuf, cbuf, NN * HID);
  k_lin1<<<gridWave, blk, 0, stream>>>(x, w1, b1, xh);

  const float* xc = xh;
  for (int l = 0; l < NLAYERS; ++l) {
    k_layer_init<<<gridNH, blk, 0, stream>>>(nmax, nsum, rst);
    k_gat_feat<<<gridWave, blk, 0, stream>>>(xc, gat_fc + (size_t)l * HID * HID, feat);
    k_el_er<<<gridNode, blk, 0, stream>>>(feat, attn_l + (size_t)l * HID,
                                          attn_r + (size_t)l * HID, el, er);
    k_edge_max<<<gridEdge, blk, 0, stream>>>(src, dst, el, er, nmax);
    k_edge_sum<<<gridEdge, blk, 0, stream>>>(src, dst, el, er, nmax, nsum);
    k_edge_aggr<<<gridEdge, blk, 0, stream>>>(src, dst, el, er, nmax, nsum, feat, rst);
    k_lstm<<<gridWave, blk, 0, stream>>>(rst, gat_bias + (size_t)l * HID, hbuf, cbuf,
                                         wih + (size_t)l * 4 * HID * HID,
                                         whh + (size_t)l * 4 * HID * HID,
                                         bih + (size_t)l * 4 * HID,
                                         bhh + (size_t)l * 4 * HID);
    xc = hbuf;
  }
  k_lin2<<<gridWave, blk, 0, stream>>>(hbuf, w2, b2, out);
}
